// CommonSpaceMultimodalLayernormRHPNet3_12446815224136
// MI455X (gfx1250) — matrix operand order repaired and checked by reading, not compiled, not run on hardware
//
#include <hip/hip_runtime.h>

// ---------------------------------------------------------------------------
// b=32, s=2048, h=128, A=8, win=3.  Outputs: attn [32,8,2048] ++ rep [32,8,128].
// Algebraic refactor:
//   C[a]   = aspProj[a] @ embedR[a]            (128 x 3 per aspect)
//   g[s,c] = X[b,s,:] . C[col=c]               (WMMA GEMM, N padded 24->32)
//   score[s,a] = g[s-1,3a] + g[s,3a+1] + g[s+1,3a+2]
//   attn = softmax_s(mask ? score : -inf)
//   Y[a,:] = sum_s attn[a,s] * X[b,s,:]        (WMMA GEMM, split-K over s)
//   rep[a,:] = Y[a,:] @ aspProj[a]             (tiny matvec)
// ---------------------------------------------------------------------------

typedef __attribute__((ext_vector_type(16))) __bf16 v16bf;
typedef __attribute__((ext_vector_type(8)))  float  v8f;

constexpr int BSZ = 32;
constexpr int SEQ = 2048;
constexpr int H   = 128;
constexpr int NA  = 8;
constexpr int NC  = 24;
constexpr int NCP = 32;
constexpr int CH  = 64;     // score rows per block
constexpr int RB  = 96;     // g rows computed (CH + 16 halo each side)
constexpr int SX  = 132;    // padded LDS stride (u16), keeps pairs 4B-aligned
constexpr int KCH = 16;     // rep: split-K chunks (2048/128)

#if defined(__has_builtin)
#  if __has_builtin(__builtin_amdgcn_global_load_async_to_lds_b128)
#    define HAS_ASYNC_LDS 1
#  endif
#endif
#ifndef HAS_ASYNC_LDS
#  define HAS_ASYNC_LDS 0
#endif

#if HAS_ASYNC_LDS
// Exact element type from the compiler diagnostic: v4i (vector_size 16).
typedef int v4i_t __attribute__((__vector_size__(4 * sizeof(int))));
typedef __attribute__((address_space(1))) v4i_t g_v4i;   // global (AS1)
typedef __attribute__((address_space(3))) v4i_t l_v4i;   // LDS (AS3)
__device__ __forceinline__ void async_copy16(const void* gp, void* lp) {
  // generic->AS1: bit-identical; generic LDS ptr low 32 bits == LDS byte offset.
  __builtin_amdgcn_global_load_async_to_lds_b128(
      (g_v4i*)(unsigned long long)(size_t)gp,
      (l_v4i*)(unsigned int)(size_t)lp, 0, 0);
}
__device__ __forceinline__ void async_wait0() {
#  if __has_builtin(__builtin_amdgcn_s_wait_asynccnt)
  __builtin_amdgcn_s_wait_asynccnt(0);
#  else
  asm volatile("s_wait_asynccnt 0" ::: "memory");
#  endif
}
#endif

__device__ __forceinline__ unsigned short f2bf(float f) {
  unsigned int u = __float_as_uint(f);
  u += 0x7FFFu + ((u >> 16) & 1u);           // round-to-nearest-even
  return (unsigned short)(u >> 16);
}
__device__ __forceinline__ __bf16 bits2bf(unsigned short u) {
  union { unsigned short s; __bf16 b; } c; c.s = u; return c.b;
}

// A operand, 16-bit 16x32 (ISA 7.12.2)
template<int STRIDE>
__device__ __forceinline__ v16bf packA(const unsigned short* m, int row, int half, int kbase) {
  v16bf a;
#pragma unroll
  for (int v = 0; v < 8; ++v) {
    int k = kbase + ((v >> 2) << 4) + (half << 3) + ((v & 3) << 1);
    unsigned int p = *(const unsigned int*)&m[row * STRIDE + k];
    a[2 * v]     = bits2bf((unsigned short)(p & 0xffffu));
    a[2 * v + 1] = bits2bf((unsigned short)(p >> 16));
  }
  return a;
}
// B operand, 16-bit 32x16, storage transposed [n][k]
template<int STRIDE>
__device__ __forceinline__ v16bf packB(const unsigned short* m, int col, int half, int kbase) {
  v16bf b;
#pragma unroll
  for (int v = 0; v < 8; ++v) {
    int k = kbase + ((v >> 2) << 4) + (half << 3) + ((v & 3) << 1);
    unsigned int p = *(const unsigned int*)&m[col * STRIDE + k];
    b[2 * v]     = bits2bf((unsigned short)(p & 0xffffu));
    b[2 * v + 1] = bits2bf((unsigned short)(p >> 16));
  }
  return b;
}

// ---- Kernel 1: C = aspProj @ embedR, bf16 transposed [NCP][H] in ws ---------
__global__ __launch_bounds__(256) void k_prep(const float* __restrict__ aspProj,
                                              const float* __restrict__ embW,
                                              unsigned short* __restrict__ Ctw) {
  int tid = threadIdx.x;
  for (int o = tid; o < NCP * H; o += 256) {
    int n = o / H, e = o % H;
    float sum = 0.f;
    if (n < NC) {
      int a = n / 3, w = n % 3;
      const float* Wp = aspProj + ((size_t)a * H + e) * H;
      const float* Ep = embW + (size_t)a * (3 * H) + w;
      for (int d = 0; d < H; ++d) sum += Wp[d] * Ep[d * 3];
    }
    Ctw[o] = f2bf(sum);
  }
}

// ---- Kernel 2: WMMA scores + window shift-combine ---------------------------
__global__ __launch_bounds__(256) void k_scores(const float* __restrict__ X,
                                                const unsigned short* __restrict__ Ctw,
                                                float* __restrict__ attn) {
#if HAS_ASYNC_LDS
  __shared__ __align__(16) float Fbuf[RB * H];    // f32 async staging, reused as g
#else
  __shared__ __align__(16) float Fbuf[RB * NCP];  // g only
#endif
  __shared__ __align__(16) unsigned short Xs[RB * SX];
  __shared__ __align__(16) unsigned short Cs[NCP * SX];
  float* g = Fbuf;

  int blk = blockIdx.x;
  int b = blk >> 5;
  int c = blk & 31;
  int s0 = c * CH;
  int base = s0 - 16;
  int tid = threadIdx.x;
  const float* Xb = X + (size_t)b * SEQ * H;

#if HAS_ASYNC_LDS
  // Async-copy the in-range f32 rows into LDS staging (ASYNCcnt-tracked),
  // overlap with the (tiny) C^T copy, then convert to bf16 in LDS.
  for (int idx = tid * 4; idx < RB * H; idx += 256 * 4) {
    int row = idx >> 7;
    int s = base + row;
    if (s >= 0 && s < SEQ)
      async_copy16(Xb + (size_t)s * H + (idx & 127), &Fbuf[idx]);
  }
  for (int idx = tid; idx < NCP * H; idx += 256) {
    int n = idx >> 7, k = idx & 127;
    Cs[n * SX + k] = Ctw[n * H + k];
  }
  async_wait0();
  __syncthreads();
  for (int idx = tid; idx < RB * H; idx += 256) {
    int row = idx >> 7;
    int s = base + row;
    float v = (s >= 0 && s < SEQ) ? Fbuf[idx] : 0.f;
    Xs[row * SX + (idx & 127)] = f2bf(v);     // zero rows realize the padding
  }
#else
  for (int idx = tid; idx < RB * H; idx += 256) {
    int row = idx >> 7, d = idx & 127;
    int s = base + row;
    float v = (s >= 0 && s < SEQ) ? Xb[(size_t)s * H + d] : 0.f;
    Xs[row * SX + d] = f2bf(v);
  }
  for (int idx = tid; idx < NCP * H; idx += 256) {
    int n = idx >> 7, k = idx & 127;
    Cs[n * SX + k] = Ctw[n * H + k];
  }
#endif
  __syncthreads();   // staging dead from here on; Fbuf reused as g

  int wave = tid >> 5, lane = tid & 31, half = lane >> 4, l15 = lane & 15;
  for (int job = wave; job < 12; job += 8) {       // 6 M-tiles x 2 N-tiles
    int mt = job >> 1, nt = job & 1;
    int mrow = mt * 16 + l15;
    int nb = nt * 16;
    v8f acc = {0.f, 0.f, 0.f, 0.f, 0.f, 0.f, 0.f, 0.f};
#pragma unroll
    for (int ks = 0; ks < 4; ++ks) {               // K = 128 = 4 x 32
      v16bf av = packA<SX>(Xs, mrow, half, ks * 32);
      v16bf bv = packB<SX>(Cs, nb + l15, half, ks * 32);
      acc = __builtin_amdgcn_wmma_f32_16x16x32_bf16(false, av, false, bv,
                                                    (short)0, acc, false, false);
    }
#pragma unroll
    for (int r = 0; r < 8; ++r)
      g[(mt * 16 + half * 8 + r) * NCP + nb + l15] = acc[r];
  }
  __syncthreads();

  for (int idx = tid; idx < CH * NA; idx += 256) {
    int sr = idx >> 3, a = idx & 7;
    int r = sr + 16;
    float sc = g[(r - 1) * NCP + a * 3 + 0]
             + g[(r    ) * NCP + a * 3 + 1]
             + g[(r + 1) * NCP + a * 3 + 2];
    attn[((size_t)b * NA + a) * SEQ + s0 + sr] = sc;
  }
}

// ---- Kernel 3: masked softmax over sequence, in place in d_out --------------
__global__ __launch_bounds__(256) void k_softmax(float* __restrict__ attn,
                                                 const unsigned char* __restrict__ mask) {
  int row = blockIdx.x;
  int b = row >> 3;
  float* p = attn + (size_t)row * SEQ;
  const unsigned char* m = mask + (size_t)b * SEQ;
  __shared__ float red[256];
  int tid = threadIdx.x;
  const float NEGINF = -__builtin_inff();

  float vals[8]; float mx = NEGINF;
#pragma unroll
  for (int i = 0; i < 8; ++i) {
    int s = tid + i * 256;
    float v = m[s] ? p[s] : NEGINF;
    vals[i] = v; mx = fmaxf(mx, v);
  }
  red[tid] = mx; __syncthreads();
  for (int off = 128; off > 0; off >>= 1) {
    if (tid < off) red[tid] = fmaxf(red[tid], red[tid + off]);
    __syncthreads();
  }
  mx = red[0]; __syncthreads();

  float sum = 0.f;
#pragma unroll
  for (int i = 0; i < 8; ++i) {
    float e = (vals[i] == NEGINF) ? 0.f : __expf(vals[i] - mx);
    vals[i] = e; sum += e;
  }
  red[tid] = sum; __syncthreads();
  for (int off = 128; off > 0; off >>= 1) {
    if (tid < off) red[tid] += red[tid + off];
    __syncthreads();
  }
  float inv = (red[0] > 0.f) ? 1.f / red[0] : 0.f;
#pragma unroll
  for (int i = 0; i < 8; ++i) p[tid + i * 256] = vals[i] * inv;
}

// ---- Kernel 4a: split-K partial Y = attn^T X (WMMA), 512 blocks -------------
__global__ __launch_bounds__(256) void k_repY(const float* __restrict__ X,
                                              const float* __restrict__ attn,
                                              float* __restrict__ partY) {
  __shared__ __align__(16) unsigned short As[16 * SX];
  __shared__ __align__(16) unsigned short Xt[H * SX];
  int bx = blockIdx.x;
  int b = bx >> 4, c = bx & 15;          // 16 k-chunks of 128 rows
  int tid = threadIdx.x;
  int wave = tid >> 5, lane = tid & 31, half = lane >> 4, l15 = lane & 15;
  int nb = wave * 16;

  for (int idx = tid; idx < 16 * 128; idx += 256) {
    int a = idx >> 7, k = idx & 127;
    float v = (a < NA) ? attn[((size_t)b * NA + a) * SEQ + c * 128 + k] : 0.f;
    As[a * SX + k] = f2bf(v);
  }
  for (int idx = tid; idx < H * 128; idx += 256) {
    int n = idx & 127, k = idx >> 7;     // coalesced over n
    Xt[n * SX + k] = f2bf(X[(size_t)b * SEQ * H + (size_t)(c * 128 + k) * H + n]);
  }
  __syncthreads();

  v8f acc = {0.f, 0.f, 0.f, 0.f, 0.f, 0.f, 0.f, 0.f};
#pragma unroll
  for (int ks = 0; ks < 4; ++ks) {
    v16bf av = packA<SX>(As, l15, half, ks * 32);
    v16bf bv = packB<SX>(Xt, nb + l15, half, ks * 32);
    acc = __builtin_amdgcn_wmma_f32_16x16x32_bf16(false, av, false, bv,
                                                  (short)0, acc, false, false);
  }
#pragma unroll
  for (int r = 0; r < 8; ++r) {
    int m = half * 8 + r;
    partY[(((size_t)bx) * 16 + m) * H + nb + l15] = acc[r];
  }
}

// ---- Kernel 4b: reduce partials + rep = Y @ aspProj -------------------------
__global__ __launch_bounds__(256) void k_repFinal(const float* __restrict__ partY,
                                                  const float* __restrict__ aspProj,
                                                  float* __restrict__ rep) {
  __shared__ float Yl[NA * H];
  int b = blockIdx.x, tid = threadIdx.x;
  for (int o = tid; o < NA * H; o += 256) {
    int a = o >> 7, d = o & 127;
    float s = 0.f;
#pragma unroll
    for (int c = 0; c < KCH; ++c)
      s += partY[(((size_t)(b * KCH + c)) * 16 + a) * H + d];
    Yl[o] = s;
  }
  __syncthreads();
  for (int o = tid; o < NA * H; o += 256) {
    int a = o >> 7, h = o & 127;
    float sum = 0.f;
    const float* Wp = aspProj + (size_t)a * H * H + h;
    for (int d = 0; d < H; ++d) sum += Yl[a * H + d] * Wp[(size_t)d * H];
    rep[((size_t)b * NA + a) * H + h] = sum;
  }
}

// ---- Kernel 4 fallback (small ws): fused per-b rep --------------------------
__global__ __launch_bounds__(256) void k_rep_fused(const float* __restrict__ X,
                                                   const float* __restrict__ attn,
                                                   const float* __restrict__ aspProj,
                                                   float* __restrict__ rep) {
  __shared__ __align__(16) unsigned short As[16 * SX];
  __shared__ __align__(16) unsigned short Xt[H * SX];
  __shared__ float Yl[16 * H];
  int b = blockIdx.x, tid = threadIdx.x;
  int wave = tid >> 5, lane = tid & 31, half = lane >> 4, l15 = lane & 15;
  int nb = wave * 16;
  v8f acc = {0.f, 0.f, 0.f, 0.f, 0.f, 0.f, 0.f, 0.f};

  for (int kc = 0; kc < KCH; ++kc) {
    __syncthreads();
    for (int idx = tid; idx < 16 * 128; idx += 256) {
      int a = idx >> 7, k = idx & 127;
      float v = (a < NA) ? attn[((size_t)b * NA + a) * SEQ + kc * 128 + k] : 0.f;
      As[a * SX + k] = f2bf(v);
    }
    for (int idx = tid; idx < H * 128; idx += 256) {
      int n = idx & 127, k = idx >> 7;
      Xt[n * SX + k] = f2bf(X[(size_t)b * SEQ * H + (size_t)(kc * 128 + k) * H + n]);
    }
    if (kc + 1 < KCH) {  // prefetch next chunk (global_prefetch_b8)
      const char* nxt = (const char*)&X[(size_t)b * SEQ * H + (size_t)(kc + 1) * 128 * H];
      __builtin_prefetch(nxt + tid * 256, 0, 0);
    }
    __syncthreads();
#pragma unroll
    for (int ks = 0; ks < 4; ++ks) {
      v16bf av = packA<SX>(As, l15, half, ks * 32);
      v16bf bv = packB<SX>(Xt, nb + l15, half, ks * 32);
      acc = __builtin_amdgcn_wmma_f32_16x16x32_bf16(false, av, false, bv,
                                                    (short)0, acc, false, false);
    }
  }
#pragma unroll
  for (int r = 0; r < 8; ++r)
    Yl[(half * 8 + r) * H + nb + l15] = acc[r];
  __syncthreads();

  for (int o = tid; o < NA * H; o += 256) {
    int a = o >> 7, h = o & 127;
    float sum = 0.f;
    const float* Wp = aspProj + (size_t)a * H * H + h;
    for (int d = 0; d < H; ++d) sum += Yl[a * H + d] * Wp[(size_t)d * H];
    rep[((size_t)b * NA + a) * H + h] = sum;
  }
}

// ---------------------------------------------------------------------------
extern "C" void kernel_launch(void* const* d_in, const int* in_sizes, int n_in,
                              void* d_out, int out_size, void* d_ws, size_t ws_size,
                              hipStream_t stream) {
  (void)in_sizes; (void)n_in; (void)out_size;
  const float*         X    = (const float*)d_in[0];
  const unsigned char* mask = (const unsigned char*)d_in[1];
  const float*         embW = (const float*)d_in[2];
  const float*         aspP = (const float*)d_in[3];

  float* attn = (float*)d_out;
  float* rep  = (float*)d_out + (size_t)BSZ * NA * SEQ;
  unsigned short* Ctw = (unsigned short*)d_ws;                   // 8 KB
  float* partY = (float*)((char*)d_ws + 8192);                   // 4 MB split-K
  const size_t needWS = 8192 + (size_t)BSZ * KCH * 16 * H * sizeof(float);

  k_prep   <<<1,        256, 0, stream>>>(aspP, embW, Ctw);
  k_scores <<<BSZ * 32, 256, 0, stream>>>(X, Ctw, attn);
  k_softmax<<<BSZ * NA, 256, 0, stream>>>(attn, mask);
  if (ws_size >= needWS) {
    k_repY    <<<BSZ * KCH, 256, 0, stream>>>(X, attn, partY);
    k_repFinal<<<BSZ,       256, 0, stream>>>(partY, aspP, rep);
  } else {
    k_rep_fused<<<BSZ,      256, 0, stream>>>(X, attn, aspP, rep);
  }
}
